// VectorQuantizer_8976481649064
// MI455X (gfx1250) — matrix operand order repaired and checked by reading, not compiled, not run on hardware
//
#include <hip/hip_runtime.h>
#include <stdint.h>

// ---------------------------------------------------------------------------
// VQ forward for MI455X (gfx1250, wave32, WMMA).
//
// Shapes: z (16,64,64,64) fp32 -> n=65536 rows of d=64; codebook (2048,64).
// Outputs (concat, fp32): q[4194304], loss, codebook_loss, commitment_loss,
//                         distance_prob[65536*2048].
//
// Roofline: ~570 MB of HBM traffic (dominated by the 536 MB prob tensor)
// => ~25us floor @ 23.3 TB/s. GEMM is only 17.2 GFLOP (inner dim 64), so we
// fuse GEMM + argmin + softmax in one pass, holding each 16x2048 fp32
// distance tile in LDS (128 KB < 320 KB/WGP) so probs are written exactly
// once, as full-wave 128B coalesced bursts (one wave per softmax row).
// Matrix op: v_wmma_f32_16x16x32_f16 (normalized operands, |x|<=1 ->
// fp16-safe), 2 WMMAs per 16x16 output tile.
// ---------------------------------------------------------------------------

typedef __attribute__((ext_vector_type(16))) _Float16 v16h;
typedef __attribute__((ext_vector_type(8)))  float    v8f;

union FragH { v16h h; uint32_t u[8]; };

#define NROWS       65536
#define KCB         2048
#define DDIM        64
#define LDS_STRIDE  2052     // 2048 + 4 dword pad -> rotate banks between rows

// -------------------- kernel 1: normalize codebook -------------------------
__global__ void vq_norm_cb(const float* __restrict__ cb,
                           float* __restrict__ cbn32,
                           _Float16* __restrict__ cb16,
                           float* __restrict__ s_c)
{
    int r = blockIdx.x * blockDim.x + threadIdx.x;
    if (r >= KCB) return;
    const float* src = cb + (size_t)r * DDIM;
    float vals[DDIM];
    float ss = 0.0f;
    #pragma unroll
    for (int j = 0; j < DDIM; ++j) { float v = src[j]; vals[j] = v; ss += v * v; }
    float inv = 1.0f / fmaxf(sqrtf(ss), 1e-12f);
    float*    d32 = cbn32 + (size_t)r * DDIM;
    _Float16* d16 = cb16  + (size_t)r * DDIM;
    #pragma unroll
    for (int j = 0; j < DDIM; ++j) {
        float vn = vals[j] * inv;
        d32[j] = vn;
        d16[j] = (_Float16)vn;
    }
    s_c[r] = ss * inv * inv;   // ||c_norm||^2 (== 1 up to rounding)
}

// -------------------- kernel 2: normalize z rows ---------------------------
// z is (b, d, h, w); row t = (b,h,w); element d at stride 4096 floats.
// Lanes take consecutive w -> all 64 loads coalesce across the wave.
__global__ __launch_bounds__(256) void vq_norm_z(const float* __restrict__ z,
                                                 _Float16* __restrict__ z16,
                                                 float* __restrict__ s_z,
                                                 float* __restrict__ inv_z)
{
    int t   = blockIdx.x * 256 + threadIdx.x;       // 0..65535
    int bb  = t >> 12;
    int rem = t & 4095;                             // h*64 + w
    size_t base = (size_t)bb * 262144 + (size_t)rem;
    float vals[DDIM];
    float ss = 0.0f;
    #pragma unroll
    for (int dd = 0; dd < DDIM; ++dd) {
        float v = z[base + (size_t)dd * 4096];
        vals[dd] = v; ss += v * v;
    }
    float inv = 1.0f / fmaxf(sqrtf(ss), 1e-12f);
    _Float16* zr = z16 + (size_t)t * DDIM;
    #pragma unroll
    for (int dd = 0; dd < DDIM; ++dd) zr[dd] = (_Float16)(vals[dd] * inv);
    inv_z[t] = inv;
    s_z[t]   = ss * inv * inv;                      // ||z_norm||^2
}

// -------- kernel 3: fused WMMA distances + argmin + softmax + q + loss -----
// Block = 256 threads (8 waves), owns 16 z-rows x all 2048 codes.
// Phase 1: each wave computes 16 N-tiles (16x16) via v_wmma_f32_16x16x32_f16,
//          converts to squared distances, parks them in the LDS tile.
// Phase 2: one full wave per row -> conflict-free LDS scans, 32-lane
//          reductions, and 128B-coalesced prob stores.
// Phase 3: row-fastest lane mapping -> 64B-coalesced q stores / z reloads.
__global__ __launch_bounds__(256) void vq_gemm_softmax(
    const float*    __restrict__ z,
    const _Float16* __restrict__ z16,
    const _Float16* __restrict__ cb16,
    const float*    __restrict__ s_z,
    const float*    __restrict__ inv_z,
    const float*    __restrict__ s_c,
    const float*    __restrict__ cbn32,
    float* __restrict__ out_q,
    float* __restrict__ out_prob,
    float* __restrict__ partials)
{
    extern __shared__ float smem[];        // [16*LDS_STRIDE] dist tile + acc
    __shared__ float szsh[16];
    __shared__ int   idxsh[16];
    float* acc = &smem[16 * LDS_STRIDE];

    const int tid = threadIdx.x;
    const int m0  = blockIdx.x * 16;       // first z-row of this block
    if (tid < 16) szsh[tid] = s_z[m0 + tid];
    if (tid == 0) *acc = 0.0f;
    __syncthreads();

    const int lane = tid & 31;
    const int wave = tid >> 5;
    const int l15  = lane & 15;
    const int hs   = lane >> 4;            // half-wave select

    // ---- A fragments: 16 rows x (K=0..31 | K=32..63) of z16, ISA layout:
    // lanes 0-15 hold K pairs {0..7,16..23}, lanes 16-31 hold {8..15,24..31}.
    FragH a0, a1;
    const uint32_t* zrow = (const uint32_t*)(z16 + (size_t)(m0 + l15) * DDIM);
    #pragma unroll
    for (int j = 0; j < 8; ++j) {
        int p = j + ((j >> 2) << 2) + (hs << 2);   // {0..3,8..11}+4*hs
        a0.u[j] = zrow[p];
        a1.u[j] = zrow[p + 16];
    }

    // ---- 16 output tiles per wave: cols [wave*256, wave*256+255] ----
    #pragma unroll 1
    for (int t = 0; t < 16; ++t) {
        const int n0  = (wave * 16 + t) * 16;
        const int col = n0 + l15;
        // B layout: lane = column (codebook entry); lanes 0-15 carry K {0..7,16..23},
        // lanes 16-31 carry K {8..15,24..31} (same order as A).
        const uint32_t* crow = (const uint32_t*)(cb16 + (size_t)col * DDIM);
        FragH b0, b1;
        #pragma unroll
        for (int j = 0; j < 8; ++j) {
            b0.u[j] = crow[j + ((j >> 2) << 2) + (hs << 2)];
            b1.u[j] = crow[16 + j + ((j >> 2) << 2) + (hs << 2)];
        }
        v8f c = {};
        c = __builtin_amdgcn_wmma_f32_16x16x32_f16(false, a0.h, false, b0.h,
                                                   (short)0, c, false, false);
        c = __builtin_amdgcn_wmma_f32_16x16x32_f16(false, a1.h, false, b1.h,
                                                   (short)0, c, false, false);
        const float sc = s_c[col];
        // C layout: VGPR v -> row v + 8*hs, col = lane&15 (within tile)
        #pragma unroll
        for (int v = 0; v < 8; ++v) {
            const int rl = v + (hs << 3);
            float dist = szsh[rl] + sc - 2.0f * c[v];
            smem[rl * LDS_STRIDE + col] = dist;
        }
    }
    __syncthreads();

    // ---- phase 2: one full wave per row (2 rows per wave) ----
    #pragma unroll 1
    for (int r = 0; r < 2; ++r) {
        const int row  = wave * 2 + r;
        const int grow = m0 + row;
        float* drow = &smem[row * LDS_STRIDE];

        float bmin = 3.4e38f;
        int   bidx = 0x7fffffff;
        for (int c = lane; c < KCB; c += 32) {
            float d = drow[c];
            if (d < bmin) { bmin = d; bidx = c; }  // ascending: first min
        }
        #pragma unroll
        for (int m = 1; m < 32; m <<= 1) {
            float om = __shfl_xor(bmin, m, 32);
            int   oi = __shfl_xor(bidx, m, 32);
            if (om < bmin || (om == bmin && oi < bidx)) { bmin = om; bidx = oi; }
        }
        if (lane == 0) idxsh[row] = bidx;

        // softmax(-dist): max of -dist is -bmin; exp(bmin - dist), sum, scale.
        float sum = 0.0f;
        for (int c = lane; c < KCB; c += 32) {
            float e = __expf(bmin - drow[c]);
            drow[c] = e;                           // stash exp in LDS
            sum += e;
        }
        #pragma unroll
        for (int m = 1; m < 32; m <<= 1) sum += __shfl_xor(sum, m, 32);
        const float invsum = 1.0f / sum;

        float* prow = out_prob + (size_t)grow * KCB;
        for (int c = lane; c < KCB; c += 32)       // 128B burst per wave/iter
            prow[c] = drow[c] * invsum;
    }
    __syncthreads();

    // ---- phase 3: q (straight-through forward) + loss ----
    // row index fastest across lanes -> rows are 16 consecutive w positions
    // at fixed (b,h,dd): 64B-coalesced q stores and z reloads.
    {
        const int row  = tid & 15;
        const int grow = m0 + row;
        const int bidx = idxsh[row];
        const int bb = grow >> 12;
        const int hh = (grow >> 6) & 63;
        const int ww = grow & 63;
        const size_t sbase = (size_t)bb * 262144 + (size_t)hh * 64 + (size_t)ww;
        const float  invn  = inv_z[grow];
        const float* qrow  = cbn32 + (size_t)bidx * DDIM;
        float part = 0.0f;
        #pragma unroll
        for (int j = 0; j < 4; ++j) {
            const int dd = (tid >> 4) + 16 * j;
            float zq = qrow[dd];
            float zn = z[sbase + (size_t)dd * 4096] * invn;
            float df = zq - zn;
            part += df * df;                       // both MSE terms' element
            out_q[sbase + (size_t)dd * 4096] = zn + (zq - zn); // z_q_st fwd
        }
        #pragma unroll
        for (int m = 1; m < 32; m <<= 1) part += __shfl_xor(part, m, 32);
        if (lane == 0) atomicAdd(acc, part);
    }

    __syncthreads();
    if (tid == 0) partials[blockIdx.x] = *acc;
}

// -------------------- kernel 4: reduce loss partials -----------------------
__global__ void vq_finalize(const float* __restrict__ partials,
                            float* __restrict__ out_scalars)
{
    __shared__ float red[256];
    int t = threadIdx.x;
    float s = 0.0f;
    for (int i = t; i < 4096; i += 256) s += partials[i];
    red[t] = s;
    __syncthreads();
    for (int k = 128; k > 0; k >>= 1) {
        if (t < k) red[t] += red[t + k];
        __syncthreads();
    }
    if (t == 0) {
        float cbl = red[0] / (65536.0f * 64.0f);
        out_scalars[0] = cbl * 1.25f;   // loss = codebook + 0.25*commitment
        out_scalars[1] = cbl;           // codebook_loss
        out_scalars[2] = cbl;           // commitment_loss (identical fwd value)
    }
}

// ---------------------------------------------------------------------------
extern "C" void kernel_launch(void* const* d_in, const int* in_sizes, int n_in,
                              void* d_out, int out_size, void* d_ws, size_t ws_size,
                              hipStream_t stream)
{
    const float* z  = (const float*)d_in[0];   // (16,64,64,64) fp32
    const float* cb = (const float*)d_in[1];   // (2048,64) fp32

    float* out         = (float*)d_out;
    float* out_q       = out;                  // 4,194,304
    float* out_scalars = out + 4194304;        // loss, cb_loss, commit_loss
    float* out_prob    = out + 4194307;        // 134,217,728

    // workspace layout (bytes)
    char* ws = (char*)d_ws;
    _Float16* z16    = (_Float16*)(ws + 0);          // 65536*64*2 = 8388608
    _Float16* cb16   = (_Float16*)(ws + 8388608);    // 2048*64*2  =  262144
    float*    s_z    = (float*)   (ws + 8650752);    // 65536*4
    float*    inv_z  = (float*)   (ws + 8912896);    // 65536*4
    float*    s_c    = (float*)   (ws + 9175040);    // 2048*4
    float*    cbn32  = (float*)   (ws + 9183232);    // 2048*64*4
    float*    partials = (float*) (ws + 9707520);    // 4096*4

    vq_norm_cb<<<8, 256, 0, stream>>>(cb, cbn32, cb16, s_c);
    vq_norm_z <<<256, 256, 0, stream>>>(z, z16, s_z, inv_z);

    const size_t shmem = (size_t)(16 * LDS_STRIDE + 1) * sizeof(float); // ~128KB
    vq_gemm_softmax<<<4096, 256, shmem, stream>>>(z, z16, cb16, s_z, inv_z, s_c,
                                                  cbn32, out_q, out_prob, partials);

    vq_finalize<<<1, 256, 0, stream>>>(partials, out_scalars);
}
